// RNNSaSLayer_2491081031812
// MI455X (gfx1250) — matrix operand order repaired and checked by reading, not compiled, not run on hardware
//
#include <hip/hip_runtime.h>
#include <hip/hip_bf16.h>
#include <math.h>

// ---------------------------------------------------------------------------
// RNN SaS layer for MI455X (gfx1250), wave32 + WMMA bf16, N-register-blocked.
//   Phase 1: precompute mu/d (transposed, bf16) for input + hidden layers.
//   Phase 2: xproj GEMM pair (32768 x 512 x 1024), 1x4 tiles/wave -> d_out.
//   Phase 3: 512 sequential step kernels (64 x 1024 x 1024 pair), 1x2
//            tiles/wave, reading h from d_out[t-1], rewriting d_out[t].
// ---------------------------------------------------------------------------

typedef __attribute__((ext_vector_type(16))) __bf16 v16bf;
typedef __attribute__((ext_vector_type(8)))  float  v8f;

#define EPS_SAS 1e-10f

// Reference harness dims: S,B,I,H = 512,64,512,1024
#define S_DIM 512
#define B_DIM 64
#define I_DIM 512
#define H_DIM 1024

// ---------------------------------------------------------------------------
// Precompute effective SaS params, store bf16 TRANSPOSED: dst[n*K + k].
// mu = (1-pi)*m ; d = (1-pi)*(chi + m*m) - mu*mu
// ---------------------------------------------------------------------------
__global__ void prep_weights(const float* __restrict__ m,
                             const float* __restrict__ pi,
                             const float* __restrict__ chi,
                             __bf16* __restrict__ muT,
                             __bf16* __restrict__ dT,
                             int K, int N) {
    int idx = blockIdx.x * blockDim.x + threadIdx.x;
    if (idx >= K * N) return;
    int k = idx / N;
    int n = idx - k * N;
    float mm = m[idx], pp = pi[idx], cc = chi[idx];
    float mu = (1.0f - pp) * mm;
    float rho = (1.0f - pp) * (cc + mm * mm);
    float d = rho - mu * mu;
    size_t t = (size_t)n * K + k;
    muT[t] = (__bf16)mu;
    dT[t]  = (__bf16)d;
}

__global__ void zero_f32(float* __restrict__ p, int n) {
    int i = blockIdx.x * blockDim.x + threadIdx.x;
    if (i < n) p[i] = 0.0f;
}

// ---------------------------------------------------------------------------
// Input layer: OUT[M,N] = X@mu + eps*sqrt((X*X)@d + EPS). M=S*B, K=I, N=H.
// Each wave computes one 16-row M-tile x NB=4 N-tiles: the A fragment (and
// its square) is built once per K-step and reused for 8 WMMAs.
// ---------------------------------------------------------------------------
#define NB_IN 4
__global__ void sas_input_gemm(const float* __restrict__ X,
                               const float* __restrict__ EPSN,
                               const __bf16* __restrict__ muT,
                               const __bf16* __restrict__ dT,
                               float* __restrict__ OUT) {
    const int K = I_DIM, N = H_DIM;
    const int lane   = threadIdx.x & 31;
    const int waveId = (blockIdx.x * blockDim.x + threadIdx.x) >> 5;
    const int groupsN = N / (16 * NB_IN);               // 16
    const int tm = waveId / groupsN;                    // 0..2047
    const int tg = waveId - tm * groupsN;
    const int m0 = tm << 4;
    const int n0 = tg * (16 * NB_IN);

    const int lane15 = lane & 15;
    const int aHi = (lane >> 4) << 3;   // 0 or 8  (C row offset)
    const int bHi = (lane >> 4) << 4;   // 0 or 16 (B-frag K sub-offset)

    const float*  xrow  = X   + (size_t)(m0 + lane15) * K;
    const __bf16* muRow = muT + (size_t)(n0 + lane15) * K + bHi;
    const __bf16* dRow  = dT  + (size_t)(n0 + lane15) * K + bHi;

    v8f accG[NB_IN] = {};
    v8f accD[NB_IN] = {};
    for (int k0 = 0; k0 < K; k0 += 32) {
        v16bf a, a2;
#pragma unroll
        for (int j = 0; j < 8; ++j) {
            float v0 = xrow[k0 + bHi + j];
            float v1 = xrow[k0 + bHi + 8 + j];
            a[j]      = (__bf16)v0;
            a[8 + j]  = (__bf16)v1;
            a2[j]     = (__bf16)(v0 * v0);
            a2[8 + j] = (__bf16)(v1 * v1);
        }
#pragma unroll
        for (int u = 0; u < NB_IN; ++u) {
            const size_t boff = (size_t)u * 16 * K + k0;
            v16bf bmu = *(const v16bf*)(muRow + boff);
            v16bf bd  = *(const v16bf*)(dRow + boff);
            accG[u] = __builtin_amdgcn_wmma_f32_16x16x32_bf16(false, a,  false, bmu,
                                                              (short)0, accG[u], false, false);
            accD[u] = __builtin_amdgcn_wmma_f32_16x16x32_bf16(false, a2, false, bd,
                                                              (short)0, accD[u], false, false);
        }
    }
#pragma unroll
    for (int u = 0; u < NB_IN; ++u) {
#pragma unroll
        for (int v = 0; v < 8; ++v) {
            int row = m0 + v + aHi;
            int col = n0 + u * 16 + lane15;
            size_t idx = (size_t)row * N + col;
            OUT[idx] = accG[u][v] + EPSN[idx] * sqrtf(accD[u][v] + EPS_SAS);
        }
    }
}

// ---------------------------------------------------------------------------
// One recurrent step: IO = tanh( h@mu_h + eps*sqrt((h*h)@d_h + EPS) + IO ).
// M=B=64, K=N=H=1024. 1x2 tiles per wave -> 128 waves (latency-friendly
// parallelism, halves h re-reads from L2 vs 1 tile/wave).
// ---------------------------------------------------------------------------
#define NB_ST 2
__global__ void sas_step(const float* __restrict__ Hprev,
                         const float* __restrict__ EPSN,
                         const __bf16* __restrict__ muT,
                         const __bf16* __restrict__ dT,
                         float* __restrict__ IO) {
    const int K = H_DIM, N = H_DIM;
    const int lane   = threadIdx.x & 31;
    const int waveId = (blockIdx.x * blockDim.x + threadIdx.x) >> 5;
    const int groupsN = N / (16 * NB_ST);               // 32
    const int tm = waveId / groupsN;                    // 0..3
    const int tg = waveId - tm * groupsN;
    const int m0 = tm << 4;
    const int n0 = tg * (16 * NB_ST);

    const int lane15 = lane & 15;
    const int aHi = (lane >> 4) << 3;
    const int bHi = (lane >> 4) << 4;

    const float*  hrow  = Hprev + (size_t)(m0 + lane15) * K;
    const __bf16* muRow = muT + (size_t)(n0 + lane15) * K + bHi;
    const __bf16* dRow  = dT  + (size_t)(n0 + lane15) * K + bHi;

    v8f accG[NB_ST] = {};
    v8f accD[NB_ST] = {};
    for (int k0 = 0; k0 < K; k0 += 32) {
        v16bf a, a2;
#pragma unroll
        for (int j = 0; j < 8; ++j) {
            float v0 = hrow[k0 + bHi + j];
            float v1 = hrow[k0 + bHi + 8 + j];
            a[j]      = (__bf16)v0;
            a[8 + j]  = (__bf16)v1;
            a2[j]     = (__bf16)(v0 * v0);
            a2[8 + j] = (__bf16)(v1 * v1);
        }
#pragma unroll
        for (int u = 0; u < NB_ST; ++u) {
            const size_t boff = (size_t)u * 16 * K + k0;
            v16bf bmu = *(const v16bf*)(muRow + boff);
            v16bf bd  = *(const v16bf*)(dRow + boff);
            accG[u] = __builtin_amdgcn_wmma_f32_16x16x32_bf16(false, a,  false, bmu,
                                                              (short)0, accG[u], false, false);
            accD[u] = __builtin_amdgcn_wmma_f32_16x16x32_bf16(false, a2, false, bd,
                                                              (short)0, accD[u], false, false);
        }
    }
#pragma unroll
    for (int u = 0; u < NB_ST; ++u) {
#pragma unroll
        for (int v = 0; v < 8; ++v) {
            int row = m0 + v + aHi;
            int col = n0 + u * 16 + lane15;
            size_t idx = (size_t)row * N + col;
            float z = accG[u][v] + EPSN[idx] * sqrtf(accD[u][v] + EPS_SAS);
            IO[idx] = tanhf(z + IO[idx]);
        }
    }
}

// ---------------------------------------------------------------------------
extern "C" void kernel_launch(void* const* d_in, const int* in_sizes, int n_in,
                              void* d_out, int out_size, void* d_ws, size_t ws_size,
                              hipStream_t stream) {
    (void)in_sizes; (void)n_in; (void)out_size; (void)ws_size;

    const float* x      = (const float*)d_in[0];
    const float* eps_in = (const float*)d_in[1];
    const float* eps_h  = (const float*)d_in[2];
    const float* m_in   = (const float*)d_in[3];
    const float* pi_in  = (const float*)d_in[4];
    const float* chi_in = (const float*)d_in[5];
    const float* m_h    = (const float*)d_in[6];
    const float* pi_h   = (const float*)d_in[7];
    const float* chi_h  = (const float*)d_in[8];
    float* out = (float*)d_out;

    // Workspace layout (6.25 MB): bf16 transposed weights + zeroed h0.
    __bf16* muT_in = (__bf16*)d_ws;
    __bf16* dT_in  = muT_in + (size_t)I_DIM * H_DIM;
    __bf16* muT_h  = dT_in  + (size_t)I_DIM * H_DIM;
    __bf16* dT_h   = muT_h  + (size_t)H_DIM * H_DIM;
    float*  h0     = (float*)(dT_h + (size_t)H_DIM * H_DIM);

    {   // effective params (bf16, transposed) -- recomputed every call
        int n = I_DIM * H_DIM;
        prep_weights<<<(n + 255) / 256, 256, 0, stream>>>(m_in, pi_in, chi_in,
                                                          muT_in, dT_in, I_DIM, H_DIM);
        n = H_DIM * H_DIM;
        prep_weights<<<(n + 255) / 256, 256, 0, stream>>>(m_h, pi_h, chi_h,
                                                          muT_h, dT_h, H_DIM, H_DIM);
        n = B_DIM * H_DIM;
        zero_f32<<<(n + 255) / 256, 256, 0, stream>>>(h0, n);
    }

    {   // xproj GEMM pair -> d_out[0 : S*B*H)
        int waves  = (S_DIM * B_DIM / 16) * (H_DIM / (16 * NB_IN));  // 32768 waves
        int blocks = waves / 8;                                      // 8 waves / block
        sas_input_gemm<<<blocks, 256, 0, stream>>>(x, eps_in, muT_in, dT_in, out);
    }

    // Sequential recurrence: h(t) overwrites xproj(t) in d_out.
    const size_t step = (size_t)B_DIM * H_DIM;
    const int stepWaves  = (B_DIM / 16) * (H_DIM / (16 * NB_ST));    // 128 waves
    const int stepBlocks = stepWaves / 8;                            // 16 blocks of 256
    for (int t = 0; t < S_DIM; ++t) {
        const float* hp = (t == 0) ? h0 : (out + (size_t)(t - 1) * step);
        sas_step<<<stepBlocks, 256, 0, stream>>>(hp, eps_h + (size_t)t * step,
                                                 muT_h, dT_h, out + (size_t)t * step);
    }

    // h_last = outputs[S-1], appended after the [S,B,H] outputs.
    hipMemcpyAsync(out + (size_t)S_DIM * step, out + (size_t)(S_DIM - 1) * step,
                   step * sizeof(float), hipMemcpyDeviceToDevice, stream);
}
